// MatchCNN_st_48292612276233
// MI455X (gfx1250) — matrix operand order repaired and checked by reading, not compiled, not run on hardware
//
#include <hip/hip_runtime.h>

typedef __attribute__((ext_vector_type(16))) _Float16 v16h;
typedef __attribute__((ext_vector_type(8)))  float    v8f;

#define LEAK 0.01f

// ---------------- prep kernels ----------------

__global__ void k_embed_gather(const float* __restrict__ table,
                               const int* __restrict__ sent,
                               _Float16* __restrict__ out, int total4, int E)
{
    int t = blockIdx.x * blockDim.x + threadIdx.x;
    if (t >= total4) return;
    int idx  = t * 4;
    int word = idx / E;
    int e    = idx - word * E;
    int row  = sent[word];
    const float4 v = *(const float4*)(table + (size_t)row * E + e);
    out[idx + 0] = (_Float16)v.x;
    out[idx + 1] = (_Float16)v.y;
    out[idx + 2] = (_Float16)v.z;
    out[idx + 3] = (_Float16)v.w;
}

__global__ void k_cast4(const float* __restrict__ src, _Float16* __restrict__ dst, int total4)
{
    int t = blockIdx.x * blockDim.x + threadIdx.x;
    if (t >= total4) return;
    int idx = t * 4;
    const float4 v = *(const float4*)(src + idx);
    dst[idx + 0] = (_Float16)v.x;
    dst[idx + 1] = (_Float16)v.y;
    dst[idx + 2] = (_Float16)v.z;
    dst[idx + 3] = (_Float16)v.w;
}

__global__ void k_castpad(const float* __restrict__ src, _Float16* __restrict__ dst,
                          int N, int K, int Npad, int Kpad)
{
    int t = blockIdx.x * blockDim.x + threadIdx.x;
    if (t >= Npad * Kpad) return;
    int n = t / Kpad, k = t - n * Kpad;
    float v = (n < N && k < K) ? src[(size_t)n * K + k] : 0.f;
    dst[t] = (_Float16)v;
}

// ---------------- fused WMMA GEMM ----------------
// D = leaky_relu(A @ W^T + bias) [* zero_gate(row)]
// One workgroup = 16 M-rows x NPW*128 columns; 8 waves x NPW tiles each.
// A is staged panel-wise (<=32 K-steps = 32KB LDS) in swizzled fragment order,
// so the MMA loop is barrier-free: ds_load_b128 x2 + NPW*(global B + wmma).
template<int N, int K, int KPAD, int NPW, int LW, int CIN, int SPLIT, int GATE, int F32OUT>
__global__ __launch_bounds__(256)
void k_gemm(const _Float16* __restrict__ srcA,   // conv input (B,LW+2,CIN) or pooled3 (B,SPLIT)
            const _Float16* __restrict__ srcA2,  // img16 (B,2048) in feats mode
            const _Float16* __restrict__ W16,    // (NPW*128, KPAD) zero-padded
            const float*    __restrict__ bias,   // (N)
            void* __restrict__ outP)             // (Mtotal,N) f16 or f32
{
    constexpr int KSTEPS = KPAD / 32;
    constexpr int PANEL  = (KSTEPS < 32) ? KSTEPS : 32;   // K-steps per LDS panel
    __shared__ __align__(32) _Float16 ldsA[PANEL * 512];
    __shared__ float rowsum[16];

    const int tid   = threadIdx.x;
    const int lane  = tid & 31;
    const int wave  = tid >> 5;
    const int mbase = blockIdx.x * 16;
    const int half  = lane >> 4;
    const int nloc  = lane & 15;

    // Stage one 16x32 A tile (global K-step ksg) into LDS slot, swizzled so each
    // lane's fragment is 32B contiguous. One aligned 32-bit f16-pair per thread.
    auto stage = [&](int ksg, int slot) {
        int idx = tid * 2;                 // 0..510, even
        int m   = idx >> 5;                // 0..15
        int k   = idx & 31;                // even
        int kg  = (ksg << 5) + k;
        unsigned pair = 0u;
        if (kg < K) {
            const _Float16* p;
            int row = mbase + m;
            if constexpr (CIN > 0) {       // conv window: win[b,l,i*CIN+c] = x[b,l+i,c]
                int b = row / LW, l = row - b * LW;
                int i = kg / CIN, c = kg - i * CIN;
                p = srcA + ((size_t)(b * (LW + 2) + l + i)) * CIN + c;
            } else {                       // feats: [pooled3 (SPLIT) | image (2048)]
                p = (kg < SPLIT) ? srcA + (size_t)row * SPLIT + kg
                                 : srcA2 + (size_t)row * 2048 + (kg - SPLIT);
            }
            pair = *(const unsigned*)p;    // 2 halfs, 4B aligned (all strides even)
        }
        int h = k >> 4;
        int j = k & 15;
        *(unsigned*)&ldsA[slot * 512 + (h * 16 + m) * 16 + j] = pair;
        if constexpr (GATE) {
            _Float16 h2[2];
            *(unsigned*)h2 = pair;
            float s = (float)h2[0] + (float)h2[1];
            if (s != 0.f) atomicAdd(&rowsum[m], s);   // ds_add_f32
        }
    };

    if (GATE && tid < 16) rowsum[tid] = 0.f;
    __syncthreads();

    // B addressing straight off the kernel arg (global addrspace):
    // lane's 16 contiguous f16 of column n; tile t offsets by 16*KPAD.
    const size_t bcol = (size_t)(wave * NPW * 16 + nloc) * KPAD + half * 16;

    v8f acc[NPW] = {};

    #pragma unroll
    for (int p0 = 0; p0 < KSTEPS; p0 += PANEL) {
        constexpr int REM = KSTEPS % PANEL;
        const int pn = (p0 + PANEL <= KSTEPS) ? PANEL : REM;

        for (int it = 0; it < pn; ++it) stage(p0 + it, it);
        __syncthreads();

        for (int ks = 0; ks < pn; ++ks) {
            v16h afrag = *(const v16h*)&ldsA[ks * 512 + lane * 16];
            const size_t koff = (size_t)((p0 + ks) << 5);
            #pragma unroll
            for (int t = 0; t < NPW; ++t) {
                v16h bfrag = *(const v16h*)(W16 + bcol + (size_t)t * 16 * KPAD + koff);
                acc[t] = __builtin_amdgcn_wmma_f32_16x16x32_f16(
                    false, afrag, false, bfrag, (short)0, acc[t], false, false);
            }
        }
        __syncthreads();                   // panel reuse fence
    }

    // Epilogue: bias + leaky_relu + zero-gate, store f16 or f32 (true N stride).
    #pragma unroll
    for (int t = 0; t < NPW; ++t) {
        int o = (wave * NPW + t) * 16 + nloc;
        if (o < N) {
            float bv = bias[o];
            #pragma unroll
            for (int r = 0; r < 8; ++r) {
                int ml = half * 8 + r;     // C/D layout: lanes>=16 hold M=8..15
                float v = acc[t][r] + bv;
                v = v > 0.f ? v : LEAK * v;
                if (GATE && rowsum[ml] == 0.f) v = 0.f;
                size_t off = (size_t)(mbase + ml) * N + o;
                if constexpr (F32OUT) ((float*)outP)[off] = v;
                else                  ((_Float16*)outP)[off] = (_Float16)v;
            }
        }
    }
}

// max_pool1d kernel=2 stride=2 over sequence dim.
__global__ void k_pool(const _Float16* __restrict__ y, _Float16* __restrict__ out,
                       int total, int Lw, int Lp, int C)
{
    int t = blockIdx.x * blockDim.x + threadIdx.x;
    if (t >= total) return;
    int c = t % C; int rest = t / C;
    int pos = rest % Lp; int b = rest / Lp;
    size_t base = ((size_t)b * Lw + 2 * pos) * C + c;
    float a = (float)y[base], d = (float)y[base + C];
    out[t] = (_Float16)(a > d ? a : d);
}

// out[b] = h[b,:] . Wo + bo
__global__ void k_final(const float* __restrict__ h, const float* __restrict__ Wo,
                        const float* __restrict__ bo, float* __restrict__ out, int B, int H)
{
    int b = blockIdx.x * blockDim.x + threadIdx.x;
    if (b >= B) return;
    float s = bo[0];
    for (int i = 0; i < H; ++i) s += h[(size_t)b * H + i] * Wo[i];
    out[b] = s;
}

// ---------------- launch ----------------
extern "C" void kernel_launch(void* const* d_in, const int* in_sizes, int n_in,
                              void* d_out, int out_size, void* d_ws, size_t ws_size,
                              hipStream_t stream)
{
    const float* image = (const float*)d_in[0];
    const int*   sent  = (const int*)  d_in[1];
    const float* table = (const float*)d_in[2];
    const float* W1 = (const float*)d_in[3];  const float* b1 = (const float*)d_in[4];
    const float* W2 = (const float*)d_in[5];  const float* b2 = (const float*)d_in[6];
    const float* W3 = (const float*)d_in[7];  const float* b3 = (const float*)d_in[8];
    const float* Wm = (const float*)d_in[9];  const float* bm = (const float*)d_in[10];
    const float* Wo = (const float*)d_in[11]; const float* bo = (const float*)d_in[12];
    float* out = (float*)d_out;

    const int B = 1024, PAD = 30, E = 512, IMG = 2048;
    const int c1 = 200, c2 = 300, c3 = 300, lin2 = 400;
    const int K1 = 1536, K2 = 600, K3 = 900, Km = 2648;
    // Npad = NPW*128 (uniform tiles/wave); Kpad = ceil32(K)
    const int Np1 = 256, Np2 = 384, Np3 = 384, Npm = 512;
    const int Kp1 = 1536, Kp2 = 608, Kp3 = 928, Kpm = 2656;

    char* base = (char*)d_ws;
    size_t off = 0;
    auto alloc = [&](size_t bytes) -> char* {
        off = (off + 255) & ~(size_t)255;
        char* p = base + off; off += bytes; return p;
    };
    _Float16* emb16 = (_Float16*)alloc((size_t)B * PAD * E * 2);
    _Float16* img16 = (_Float16*)alloc((size_t)B * IMG * 2);
    _Float16* W1p   = (_Float16*)alloc((size_t)Np1 * Kp1 * 2);
    _Float16* W2p   = (_Float16*)alloc((size_t)Np2 * Kp2 * 2);
    _Float16* W3p   = (_Float16*)alloc((size_t)Np3 * Kp3 * 2);
    _Float16* Wmp   = (_Float16*)alloc((size_t)Npm * Kpm * 2);
    _Float16* y1    = (_Float16*)alloc((size_t)B * 28 * c1 * 2);
    _Float16* p1    = (_Float16*)alloc((size_t)B * 14 * c1 * 2);
    _Float16* y2    = (_Float16*)alloc((size_t)B * 12 * c2 * 2);
    _Float16* p2    = (_Float16*)alloc((size_t)B * 6  * c2 * 2);
    _Float16* y3    = (_Float16*)alloc((size_t)B * 4  * c3 * 2);
    _Float16* p3    = (_Float16*)alloc((size_t)B * 2  * c3 * 2);
    float*    hbuf  = (float*)   alloc((size_t)B * lin2 * 4);

    const int T = 256;
    auto blks = [](int n, int t) { return (n + t - 1) / t; };

    // prep: gather + casts
    int g4 = (B * PAD * E) / 4;
    k_embed_gather<<<blks(g4, T), T, 0, stream>>>(table, sent, emb16, g4, E);
    int i4 = (B * IMG) / 4;
    k_cast4<<<blks(i4, T), T, 0, stream>>>(image, img16, i4);
    k_castpad<<<blks(Np1 * Kp1, T), T, 0, stream>>>(W1, W1p, c1, K1, Np1, Kp1);
    k_castpad<<<blks(Np2 * Kp2, T), T, 0, stream>>>(W2, W2p, c2, K2, Np2, Kp2);
    k_castpad<<<blks(Np3 * Kp3, T), T, 0, stream>>>(W3, W3p, c3, K3, Np3, Kp3);
    k_castpad<<<blks(Npm * Kpm, T), T, 0, stream>>>(Wm, Wmp, lin2, Km, Npm, Kpm);

    // block 1: (B*28,1536)x(1536,200) -> y1 -> pool -> p1
    k_gemm<200, 1536, 1536, 2, 28, 512, 0, 1, 0>
        <<<(B * 28) / 16, T, 0, stream>>>(emb16, nullptr, W1p, b1, y1);
    int t1 = B * 14 * c1;
    k_pool<<<blks(t1, T), T, 0, stream>>>(y1, p1, t1, 28, 14, c1);

    // block 2: (B*12,600)x(600,300)
    k_gemm<300, 600, 608, 3, 12, 200, 0, 1, 0>
        <<<(B * 12) / 16, T, 0, stream>>>(p1, nullptr, W2p, b2, y2);
    int t2 = B * 6 * c2;
    k_pool<<<blks(t2, T), T, 0, stream>>>(y2, p2, t2, 12, 6, c2);

    // block 3: (B*4,900)x(900,300)
    k_gemm<300, 900, 928, 3, 4, 300, 0, 1, 0>
        <<<(B * 4) / 16, T, 0, stream>>>(p2, nullptr, W3p, b3, y3);
    int t3 = B * 2 * c3;
    k_pool<<<blks(t3, T), T, 0, stream>>>(y3, p3, t3, 4, 2, c3);

    // Wm: feats (B,2648) x (2648,400) -> h (f32, no gate)
    k_gemm<400, 2648, 2656, 4, 1, 0, 600, 0, 1>
        <<<B / 16, T, 0, stream>>>(p3, img16, Wmp, bm, hbuf);

    // Wo: (B,400) . (400,) + bo
    k_final<<<blks(B, T), T, 0, stream>>>(hbuf, Wo, bo, out, B, lin2);
}
